// GNNModel_81106162417980
// MI455X (gfx1250) — matrix operand order repaired and checked by reading, not compiled, not run on hardware
//
#include <hip/hip_runtime.h>
#include <hip/hip_bf16.h>

// ---------------------------------------------------------------------------
// 3-layer GAT forward for MI455X (gfx1250, wave32).
//  - feat = h @ W via v_wmma_f32_16x16x32_f16 (f16 in, f32 acc)
//  - segment max via order-preserving-uint GLOBAL_ATOMIC_MAX_U32
//  - segment sums via GLOBAL_ATOMIC_ADD_F32 (rst fits in 192MB L2)
// ---------------------------------------------------------------------------

#define N_NODES 50000
#define N_EDGES 800000
#define NEG_SLOPE 0.2f

typedef __attribute__((ext_vector_type(16))) _Float16 v16h;
typedef __attribute__((ext_vector_type(8)))  float    v8f;

// ---------------- conversion kernels ----------------

__global__ void k_f32_to_f16(const float* __restrict__ x, _Float16* __restrict__ y, long n) {
    long i = (long)blockIdx.x * blockDim.x + threadIdx.x;
    if (i < n) y[i] = (_Float16)x[i];
}

// W [K, Nout] fp32  ->  Wt [Nout, K] f16  (so B-fragment K-pairs are contiguous)
__global__ void k_w_to_f16T(const float* __restrict__ W, _Float16* __restrict__ Wt,
                            int K, int Nout) {
    int i = blockIdx.x * blockDim.x + threadIdx.x;
    if (i < K * Nout) {
        int k = i / Nout, n = i % Nout;
        Wt[(size_t)n * K + k] = (_Float16)W[i];
    }
}

// ---------------- WMMA GEMM: C[M,Nout] = A[M,K] * Bt[Nout,K]^T ----------------
// One 16x16 output tile per wave. M, Nout multiples of 16; K multiple of 32.

__global__ __launch_bounds__(128) void k_gemm_wmma(
    const _Float16* __restrict__ A, const _Float16* __restrict__ Bt,
    float* __restrict__ C, int M, int K, int Nout) {
    const int lane = threadIdx.x & 31;
    const int wave = threadIdx.x >> 5;
    const int half = lane >> 4;     // 0: lanes 0-15, 1: lanes 16-31
    const int l15  = lane & 15;
    const int nTiles = Nout >> 4;
    const int tile = blockIdx.x * 4 + wave;            // wave-uniform
    if (tile >= (M >> 4) * nTiles) return;             // wave-uniform guard
    const int mBase = (tile / nTiles) << 4;
    const int nBase = (tile % nTiles) << 4;

    const _Float16* ap = A  + (size_t)(mBase + l15) * K;  // A row, lane = M
    const _Float16* bp = Bt + (size_t)(nBase + l15) * K;  // B col, lane = N

    v8f acc = {};
    for (int kb = 0; kb < K; kb += 32) {
        v16h af, bf;
#pragma unroll
        for (int i = 0; i < 8; i++) {
            // A 16x32 f16 layout: VGPR i holds K = (i>>2)*16 + half*8 + (i&3)*2 , +1
            int ka = kb + ((i >> 2) << 4) + (half << 3) + ((i & 3) << 1);
            af[2 * i]     = ap[ka];
            af[2 * i + 1] = ap[ka + 1];
            // B 32x16 f16 layout: same K mapping as A: VGPR i holds K = (i>>2)*16 + half*8 + (i&3)*2 , +1
            int kbx = ka;
            bf[2 * i]     = bp[kbx];
            bf[2 * i + 1] = bp[kbx + 1];
        }
        acc = __builtin_amdgcn_wmma_f32_16x16x32_f16(
            /*neg_a=*/false, af, /*neg_b=*/false, bf,
            /*c_mod=*/(short)0, acc, /*reuse_a=*/false, /*reuse_b=*/false);
    }
#pragma unroll
    for (int i = 0; i < 8; i++) {
        // C/D layout: VGPR i -> M = i + 8*half, N = lane&15
        int m = mBase + i + (half << 3);
        C[(size_t)m * Nout + nBase + l15] = acc[i];
    }
}

// ---------------- attention coefficients: el/er [N,H] ----------------

__global__ void k_attn_coef(const float* __restrict__ feat, const float* __restrict__ al,
                            const float* __restrict__ ar, float* __restrict__ el,
                            float* __restrict__ er, int H, int D) {
    int i = blockIdx.x * blockDim.x + threadIdx.x;   // over N*H
    if (i >= N_NODES * H) return;
    int n = i / H, h = i % H;
    const float* f = feat + (size_t)n * H * D + (size_t)h * D;
    float sl = 0.f, sr = 0.f;
    for (int d = 0; d < D; d++) {
        float v = f[d];
        sl += v * al[h * D + d];
        sr += v * ar[h * D + d];
    }
    el[i] = sl;
    er[i] = sr;
}

// ---------------- per-layer init: m = enc(-inf), s = 0, rst = 0 ----------------

__global__ void k_init(unsigned* __restrict__ m_enc, float* __restrict__ s,
                       float* __restrict__ rst, int H, int Nout) {
    long i = (long)blockIdx.x * blockDim.x + threadIdx.x;
    if (i >= (long)N_NODES * Nout) return;
    rst[i] = 0.f;
    if (i < (long)N_NODES * H) {
        m_enc[i] = 0x007FFFFFu;   // encode(-inf) in order-preserving uint space
        s[i] = 0.f;
    }
}

// order-preserving float<->uint encoding (monotone under unsigned compare)
__device__ __forceinline__ unsigned enc_f(float v) {
    unsigned u = __float_as_uint(v);
    return (u & 0x80000000u) ? ~u : (u | 0x80000000u);
}
__device__ __forceinline__ float dec_f(unsigned u) {
    unsigned bits = (u & 0x80000000u) ? (u & 0x7FFFFFFFu) : ~u;
    return __uint_as_float(bits);
}

// ---------------- edge pass 1: e = lrelu(el[src]+er[dst]); atomicMax m[dst] ----

__global__ void k_edge_lrelu_max(const int* __restrict__ src, const int* __restrict__ dst,
                                 const float* __restrict__ el, const float* __restrict__ er,
                                 float* __restrict__ ev, unsigned* __restrict__ m_enc, int H) {
    long i = (long)blockIdx.x * blockDim.x + threadIdx.x;  // over E*H
    if (i >= (long)N_EDGES * H) return;
    long e = i / H;
    int h = (int)(i % H);
    int sN = src[e], dN = dst[e];
    float v = el[sN * H + h] + er[dN * H + h];
    v = v > 0.f ? v : NEG_SLOPE * v;
    ev[i] = v;
    atomicMax(&m_enc[dN * H + h], enc_f(v));   // -> global_atomic_max_u32
}

// ---------------- edge pass 2: ee = exp(e - m[dst]); s[dst] += ee -------------

__global__ void k_edge_exp_sum(const int* __restrict__ dst, float* __restrict__ ev,
                               const unsigned* __restrict__ m_enc, float* __restrict__ s,
                               int H) {
    long i = (long)blockIdx.x * blockDim.x + threadIdx.x;
    if (i >= (long)N_EDGES * H) return;
    long e = i / H;
    int h = (int)(i % H);
    int dN = dst[e];
    float m = dec_f(m_enc[dN * H + h]);
    float ee = __expf(ev[i] - m);
    ev[i] = ee;
    atomicAdd(&s[dN * H + h], ee);
}

// ---------------- edge pass 3: rst[dst] += feat[src] * (ee / s[dst]) ----------
// Nout/4 lanes per edge, float4 rows; atomics land in L2 (rst = 25.6MB << 192MB).

__global__ __launch_bounds__(256) void k_edge_msg(
    const int* __restrict__ src, const int* __restrict__ dst,
    const float* __restrict__ feat, const float* __restrict__ ev,
    const float* __restrict__ s, float* __restrict__ rst, int H, int D, int Nout) {
    const int TPE = Nout >> 2;            // threads per edge (float4 granules)
    const int epb = 256 / TPE;            // edges per block
    int t = threadIdx.x;
    int eLocal = t / TPE, c4 = t % TPE;
    long e = (long)blockIdx.x * epb + eLocal;
    if (e >= N_EDGES) return;
    int sN = src[e], dN = dst[e];
    int c = c4 << 2;
    int h = c / D;
    float coef = ev[e * H + h] / s[dN * H + h];
    const float* f = feat + (size_t)sN * Nout + c;
    float* r = rst + (size_t)dN * Nout + c;
    __builtin_prefetch(f, 0, 1);          // global_prefetch_b8
    float4 v = *(const float4*)f;
    atomicAdd(&r[0], v.x * coef);
    atomicAdd(&r[1], v.y * coef);
    atomicAdd(&r[2], v.z * coef);
    atomicAdd(&r[3], v.w * coef);
}

// ---------------- finalize: out = relu(rst + b) ----------------

__global__ void k_finalize(const float* __restrict__ rst, const float* __restrict__ b,
                           float* __restrict__ out, int Nout) {
    long i = (long)blockIdx.x * blockDim.x + threadIdx.x;
    if (i >= (long)N_NODES * Nout) return;
    float v = rst[i] + b[i % Nout];
    out[i] = v > 0.f ? v : 0.f;
}

// ---------------- host-side layer driver (all launches on `stream`) ----------

static void gat_layer_launch(const float* hin, int Kin, const float* W, const float* al,
                             const float* ar, const float* b, int H, int D,
                             const int* src, const int* dst,
                             _Float16* A16, _Float16* Wt16, float* feat, float* el,
                             float* er, unsigned* m_enc, float* s, float* ev,
                             float* rst, float* hout, hipStream_t stream) {
    const int Nout = H * D;
    const long nA = (long)N_NODES * Kin;
    k_f32_to_f16<<<(unsigned)((nA + 255) / 256), 256, 0, stream>>>(hin, A16, nA);
    const int nW = Kin * Nout;
    k_w_to_f16T<<<(nW + 255) / 256, 256, 0, stream>>>(W, Wt16, Kin, Nout);
    const int tiles = (N_NODES / 16) * (Nout / 16);
    k_gemm_wmma<<<(tiles + 3) / 4, 128, 0, stream>>>(A16, Wt16, feat, N_NODES, Kin, Nout);
    const int nNH = N_NODES * H;
    k_attn_coef<<<(nNH + 255) / 256, 256, 0, stream>>>(feat, al, ar, el, er, H, D);
    const long nR = (long)N_NODES * Nout;
    k_init<<<(unsigned)((nR + 255) / 256), 256, 0, stream>>>(m_enc, s, rst, H, Nout);
    const long nEH = (long)N_EDGES * H;
    k_edge_lrelu_max<<<(unsigned)((nEH + 255) / 256), 256, 0, stream>>>(src, dst, el, er, ev,
                                                                        m_enc, H);
    k_edge_exp_sum<<<(unsigned)((nEH + 255) / 256), 256, 0, stream>>>(dst, ev, m_enc, s, H);
    const int epb = 256 / (Nout >> 2);
    k_edge_msg<<<(N_EDGES + epb - 1) / epb, 256, 0, stream>>>(src, dst, feat, ev, s, rst,
                                                              H, D, Nout);
    k_finalize<<<(unsigned)((nR + 255) / 256), 256, 0, stream>>>(rst, b, hout, Nout);
}

extern "C" void kernel_launch(void* const* d_in, const int* in_sizes, int n_in,
                              void* d_out, int out_size, void* d_ws, size_t ws_size,
                              hipStream_t stream) {
    const float* x   = (const float*)d_in[0];   // [50000, 256]
    const int*   src = (const int*)d_in[1];     // [800000]
    const int*   dst = (const int*)d_in[2];     // [800000]
    const float* W0  = (const float*)d_in[3];
    const float* al0 = (const float*)d_in[4];
    const float* ar0 = (const float*)d_in[5];
    const float* b0  = (const float*)d_in[6];
    const float* W1  = (const float*)d_in[7];
    const float* al1 = (const float*)d_in[8];
    const float* ar1 = (const float*)d_in[9];
    const float* b1  = (const float*)d_in[10];
    const float* W2  = (const float*)d_in[11];
    const float* al2 = (const float*)d_in[12];
    const float* ar2 = (const float*)d_in[13];
    const float* b2  = (const float*)d_in[14];

    // workspace carve-out (256B aligned)
    char* ws = (char*)d_ws;
    size_t o = 0;
    auto carve = [&](size_t bytes) -> char* {
        char* p = ws + o;
        o += (bytes + 255) & ~(size_t)255;
        return p;
    };
    _Float16* A16  = (_Float16*)carve((size_t)N_NODES * 256 * 2);
    _Float16* Wt16 = (_Float16*)carve((size_t)256 * 128 * 2);
    float* feat  = (float*)carve((size_t)N_NODES * 128 * 4);
    float* rst   = (float*)carve((size_t)N_NODES * 128 * 4);
    float* h1    = (float*)carve((size_t)N_NODES * 128 * 4);
    float* h2    = (float*)carve((size_t)N_NODES * 128 * 4);
    float* el    = (float*)carve((size_t)N_NODES * 2 * 4);
    float* er    = (float*)carve((size_t)N_NODES * 2 * 4);
    unsigned* mE = (unsigned*)carve((size_t)N_NODES * 2 * 4);
    float* s     = (float*)carve((size_t)N_NODES * 2 * 4);
    float* ev    = (float*)carve((size_t)N_EDGES * 2 * 4);
    (void)ws_size; (void)in_sizes; (void)n_in; (void)out_size;

    // layer 0: 256 -> (2,64), relu
    gat_layer_launch(x,  256, W0, al0, ar0, b0, 2, 64, src, dst,
                     A16, Wt16, feat, el, er, mE, s, ev, rst, h1, stream);
    // layer 1: 128 -> (2,64), relu
    gat_layer_launch(h1, 128, W1, al1, ar1, b1, 2, 64, src, dst,
                     A16, Wt16, feat, el, er, mE, s, ev, rst, h2, stream);
    // layer 2: 128 -> (1,64), relu -> d_out
    gat_layer_launch(h2, 128, W2, al2, ar2, b2, 1, 64, src, dst,
                     A16, Wt16, feat, el, er, mE, s, ev, rst, (float*)d_out, stream);
}
